// TaskConditionedDSC_48387101557266
// MI455X (gfx1250) — matrix operand order repaired and checked by reading, not compiled, not run on hardware
//
#include <hip/hip_runtime.h>
#include <hip/hip_bf16.h>

// ---------------------------------------------------------------------------
// Problem constants (from reference): B=32, D=256, H=W=64, N=4096, K=6
// ---------------------------------------------------------------------------
#define NB   32
#define DD   256
#define NN   4096
#define KK   6

typedef __attribute__((ext_vector_type(16))) _Float16 v16h;
typedef __attribute__((ext_vector_type(8)))  _Float16 v8h;
typedef __attribute__((ext_vector_type(4)))  _Float16 v4h;
typedef __attribute__((ext_vector_type(8)))  float    v8f;
typedef int i4v __attribute__((__vector_size__(16)));   // int4, matches builtin proto

#define AS1 __attribute__((address_space(1)))
#define AS3 __attribute__((address_space(3)))

// workspace layout (bytes)
constexpr size_t WS_H1     = 0;                       // 32*256 f32
constexpr size_t WS_CLUE   = 32768;                   // 32*1536 f32
constexpr size_t WS_WPACK  = 229376;                  // 2*65536 f16
constexpr size_t WS_STATE  = 491520;                  // 32*256 f32
constexpr size_t WS_MASK   = 524288;                  // 32*4096 f32
constexpr size_t WS_Q      = 1048576;                 // 32*256 f32
constexpr size_t WS_SCORES = 1081344;                 // 32*4096 f32
constexpr size_t WS_ATTN   = 1605632;                 // 32*4096 f32
constexpr size_t WS_ATT    = 2129920;                 // 32*256 f32 (attended)
constexpr size_t WS_ENT    = 2162688;                 // 32 f32
constexpr size_t WS_KP     = 4194304;                 // 32*256*4096 f16 (b,d,n)
constexpr size_t WS_VP     = WS_KP + 67108864;        // same size

// d_out layout: centroids [0,384) | maps [384, 384+786432) | stops [786816, 787008)
constexpr int OUT_MAPS = 384;
constexpr int OUT_STOP = 786816;

__device__ __forceinline__ float gelu_exact(float x) {
    return 0.5f * x * (1.0f + erff(x * 0.70710678118654752f));
}

// 256-thread block reductions (wave32)
__device__ __forceinline__ float bsum256(float v) {
    #pragma unroll
    for (int o = 16; o > 0; o >>= 1) v += __shfl_down(v, o, 32);
    __shared__ float w[8];
    int lane = threadIdx.x & 31, wid = threadIdx.x >> 5;
    if (lane == 0) w[wid] = v;
    __syncthreads();
    if (wid == 0) {
        float x = (lane < 8) ? w[lane] : 0.0f;
        #pragma unroll
        for (int o = 4; o > 0; o >>= 1) x += __shfl_down(x, o, 32);
        if (lane == 0) w[0] = x;
    }
    __syncthreads();
    float r = w[0];
    __syncthreads();
    return r;
}

__device__ __forceinline__ float bmax256(float v) {
    #pragma unroll
    for (int o = 16; o > 0; o >>= 1) v = fmaxf(v, __shfl_down(v, o, 32));
    __shared__ float w[8];
    int lane = threadIdx.x & 31, wid = threadIdx.x >> 5;
    if (lane == 0) w[wid] = v;
    __syncthreads();
    if (wid == 0) {
        float x = (lane < 8) ? w[lane] : -3.4e38f;
        #pragma unroll
        for (int o = 4; o > 0; o >>= 1) x = fmaxf(x, __shfl_down(x, o, 32));
        if (lane == 0) w[0] = x;
    }
    __syncthreads();
    float r = w[0];
    __syncthreads();
    return r;
}

// ---------------------------------------------------------------------------
// init: state = 0, mask = 1
// ---------------------------------------------------------------------------
__global__ __launch_bounds__(256) void k_init(float* state, float* mask) {
    int i = blockIdx.x * 256 + threadIdx.x;          // grid covers 131072
    if (i < NB * DD) state[i] = 0.0f;
    if (i < NB * NN) mask[i] = 1.0f;
}

// ---------------------------------------------------------------------------
// context MLP stage 1: h1 = gelu(tc @ ctx_w1^T + b1)
// ---------------------------------------------------------------------------
__global__ __launch_bounds__(256) void k_ctx1(const float* __restrict__ tc,
                                              const float* __restrict__ w1,
                                              const float* __restrict__ b1,
                                              float* __restrict__ h1) {
    int b = blockIdx.x, o = threadIdx.x;
    __shared__ float xs[DD];
    xs[o] = tc[b * DD + o];
    __syncthreads();
    float a = b1[o];
    for (int d = 0; d < DD; ++d) a += xs[d] * w1[o * DD + d];
    h1[b * DD + o] = gelu_exact(a);
}

// context MLP stage 2: clue = clip(h1 @ ctx_w2^T + b2, +-10)
__global__ __launch_bounds__(256) void k_ctx2(const float* __restrict__ h1,
                                              const float* __restrict__ w2,
                                              const float* __restrict__ b2,
                                              float* __restrict__ clue) {
    int b = blockIdx.x;
    int o = blockIdx.y * 256 + threadIdx.x;          // 0..1535
    __shared__ float xs[DD];
    xs[threadIdx.x] = h1[b * DD + threadIdx.x];
    __syncthreads();
    float a = b2[o];
    for (int d = 0; d < DD; ++d) a += xs[d] * w2[o * DD + d];
    clue[b * (KK * DD) + o] = fminf(fmaxf(a, -10.0f), 10.0f);
}

// ---------------------------------------------------------------------------
// pack k_w / v_w into WMMA B-fragment order (f16), 16 halfs per lane slot.
// B 32x16 f16 order used here (A uses the same k order): n = lane%16, k = (lane<16?0:16)+j, j=0..15
// ---------------------------------------------------------------------------
__global__ __launch_bounds__(256) void k_pack(const float* __restrict__ k_w,
                                              const float* __restrict__ v_w,
                                              _Float16* __restrict__ wpack) {
    int idx  = blockIdx.x * 256 + threadIdx.x;       // 0..8191
    int lane = idx & 31;
    int nt   = (idx >> 5) & 15;
    int kk   = (idx >> 9) & 7;
    int mat  = idx >> 12;
    const float* w = mat ? v_w : k_w;
    int o  = nt * 16 + (lane & 15);
    int d0 = kk * 32 + ((lane >> 4) * 16);
    _Float16* dst = wpack + (size_t)mat * 65536 + (size_t)((kk * 16 + nt) * 32 + lane) * 16;
    #pragma unroll
    for (int j = 0; j < 16; ++j) dst[j] = (_Float16)w[o * DD + d0 + j];
}

// ---------------------------------------------------------------------------
// Fused LayerNorm + K/V projection, WMMA f16->f32.
// Tile: 32 spatial rows x D=256. Output kp/vp f16 in (b, d, n) layout.
// Feature tile staged via ASYNCcnt-tracked global->LDS copies when available.
// ---------------------------------------------------------------------------
#define TM 32
__global__ __launch_bounds__(256) void k_lnkv(const float* __restrict__ feat,
                                              const float* __restrict__ fn_g,
                                              const float* __restrict__ fn_b,
                                              const _Float16* __restrict__ wpack,
                                              const float* __restrict__ k_b,
                                              const float* __restrict__ v_b,
                                              _Float16* __restrict__ kp,
                                              _Float16* __restrict__ vp) {
    __shared__ float    raw[DD * TM];      // [d][n]  32 KB
    __shared__ _Float16 nrm[TM * DD];      // [n][d]  16 KB
    __shared__ float    ps[8][TM], ps2[8][TM];
    __shared__ float    mu[TM], rsd[TM];
    __shared__ float    g_s[DD], b_s[DD];

    int tid = threadIdx.x;
    int b   = blockIdx.x >> 7;                       // 4096/32 = 128 tiles per b
    int n0  = (blockIdx.x & 127) * TM;
    const float* src = feat + (size_t)b * DD * NN + n0;

#if __has_builtin(__builtin_amdgcn_global_load_async_to_lds_b128)
    // async copy: 256 threads x 8 iters x 16B = 32 KB tile (ASYNCcnt-tracked)
    #pragma unroll
    for (int it = 0; it < 8; ++it) {
        int i   = it * 256 + tid;                    // 0..2047 (16B chunk id)
        int d   = i >> 3;
        int sub = i & 7;
        __builtin_amdgcn_global_load_async_to_lds_b128(
            (AS1 i4v*)(src + (size_t)d * NN + sub * 4),
            (AS3 i4v*)(&raw[d * TM + sub * 4]), 0, 0);
    }
    g_s[tid] = fn_g[tid];
    b_s[tid] = fn_b[tid];
#if __has_builtin(__builtin_amdgcn_s_wait_asynccnt)
    __builtin_amdgcn_s_wait_asynccnt(0);
#else
    asm volatile("s_wait_asynccnt 0" ::: "memory");
#endif
#else
    // fallback: coalesced transpose load through VGPRs
    for (int i = tid; i < DD * TM; i += 256) {
        int d = i >> 5, n = i & 31;
        raw[i] = src[(size_t)d * NN + n];
    }
    g_s[tid] = fn_g[tid];
    b_s[tid] = fn_b[tid];
#endif
    __syncthreads();

    // layernorm over d per spatial row
    int r = tid & 31, part = tid >> 5;               // 8 partials of 32 d's
    float s = 0.0f, s2 = 0.0f;
    for (int d = part * 32; d < part * 32 + 32; ++d) {
        float x = raw[d * TM + r];
        s += x; s2 += x * x;
    }
    ps[part][r] = s; ps2[part][r] = s2;
    __syncthreads();
    if (part == 0) {
        float S = 0.0f, S2 = 0.0f;
        #pragma unroll
        for (int p = 0; p < 8; ++p) { S += ps[p][r]; S2 += ps2[p][r]; }
        float m   = S * (1.0f / 256.0f);
        float var = S2 * (1.0f / 256.0f) - m * m;
        mu[r] = m; rsd[r] = rsqrtf(var + 1e-5f);
    }
    __syncthreads();
    {
        float m = mu[r], ir = rsd[r];
        for (int d = part * 32; d < part * 32 + 32; ++d) {
            float x = (raw[d * TM + r] - m) * ir * g_s[d] + b_s[d];
            nrm[r * DD + d] = (_Float16)x;
        }
    }
    __syncthreads();

    // GEMM: wave -> (matrix, 16-row tile, 8 col tiles), B loads software-pipelined
    int wave = tid >> 5, lane = tid & 31;
    int mat  = wave >> 2;                            // 0 = K proj, 1 = V proj
    int m0   = ((wave >> 1) & 1) * 16;
    int ntb  = (wave & 1) * 8;
    const _Float16* wp   = wpack + (size_t)mat * 65536 + (size_t)lane * 16;
    const float*    bias = mat ? v_b : k_b;

    v8f acc[8] = {};
    int mrow  = m0 + (lane & 15);
    int khalf = (lane >> 4) * 16;                    // A-layout K-half selector

    auto loadB = [&](int f) -> v16h {                // f = kc*8 + j
        int kc = f >> 3, j = f & 7;
        const _Float16* bp = wp + (size_t)(kc * 16 + ntb + j) * 512;
        union { v16h v; v8h h[2]; } u;
        u.h[0] = *(const v8h*)bp;
        u.h[1] = *(const v8h*)(bp + 8);
        return u.v;
    };
    union { v16h v; v8h h[2]; } au;
    auto loadA = [&](int kc) {
        const _Float16* ap = &nrm[mrow * DD + kc * 32 + khalf];
        au.h[0] = *(const v8h*)ap;                   // K = kc*32 + khalf + 0..7
        au.h[1] = *(const v8h*)(ap + 8);             // K = kc*32 + khalf + 8..15
    };

    loadA(0);
    v16h bcur = loadB(0);
    #pragma unroll
    for (int f = 0; f < 64; ++f) {
        v16h bnext = (f < 63) ? loadB(f + 1) : bcur; // prefetch next fragment
        acc[f & 7] = __builtin_amdgcn_wmma_f32_16x16x32_f16(
            false, au.v, false, bcur, (short)0, acc[f & 7], false, false);
        if ((f & 7) == 7 && f < 63) loadA((f >> 3) + 1);
        bcur = bnext;
    }

    // store C: M = rr + 8*(lane>=16), N = lane&15; output layout (b, o, n)
    _Float16* dst = mat ? vp : kp;
    int nbase = n0 + m0 + ((lane >> 4) * 8);
    #pragma unroll
    for (int j = 0; j < 8; ++j) {
        int o = (ntb + j) * 16 + (lane & 15);
        float bo = bias[o];
        v8h outv;
        #pragma unroll
        for (int rr = 0; rr < 8; ++rr) outv[rr] = (_Float16)(acc[j][rr] + bo);
        *(v8h*)(dst + (size_t)(b * DD + o) * NN + nbase) = outv;
    }
}

// ---------------------------------------------------------------------------
// step: query = LN(clue_t + state); q = query @ q_w^T + q_b
// ---------------------------------------------------------------------------
__global__ __launch_bounds__(256) void k_query(const float* __restrict__ clue,
                                               const float* __restrict__ state,
                                               const float* __restrict__ qn_g,
                                               const float* __restrict__ qn_b,
                                               const float* __restrict__ q_w,
                                               const float* __restrict__ q_b,
                                               float* __restrict__ qv, int t) {
    int b = blockIdx.x, tid = threadIdx.x;
    __shared__ float xs[DD];
    float x  = clue[b * (KK * DD) + t * DD + tid] + state[b * DD + tid];
    float S  = bsum256(x);
    float S2 = bsum256(x * x);
    float m  = S * (1.0f / 256.0f);
    float var = S2 * (1.0f / 256.0f) - m * m;
    xs[tid] = (x - m) * rsqrtf(var + 1e-5f) * qn_g[tid] + qn_b[tid];
    __syncthreads();
    float a = q_b[tid];
    for (int d = 0; d < DD; ++d) a += xs[d] * q_w[tid * DD + d];
    qv[b * DD + tid] = a;
}

// ---------------------------------------------------------------------------
// step: masked, clipped, temperature-scaled logits (4 n's / thread, 8B loads)
// ---------------------------------------------------------------------------
__global__ __launch_bounds__(256) void k_scores(const _Float16* __restrict__ kp,
                                                const float* __restrict__ qv,
                                                const float* __restrict__ mask,
                                                float* __restrict__ scores) {
    int b  = blockIdx.x >> 2;
    int nb = (blockIdx.x & 3) * 1024 + threadIdx.x * 4;
    __shared__ float qs[DD];
    qs[threadIdx.x] = qv[b * DD + threadIdx.x];
    __syncthreads();
    const _Float16* base = kp + (size_t)b * DD * NN + nb;
    float s[4] = {0.0f, 0.0f, 0.0f, 0.0f};
    for (int d = 0; d < DD; ++d) {
        if ((d & 15) == 0) __builtin_prefetch(base + (size_t)(d + 32) * NN, 0, 0);
        v4h k4 = *(const v4h*)(base + (size_t)d * NN);
        float q = qs[d];
        #pragma unroll
        for (int j = 0; j < 4; ++j) s[j] += q * (float)k4[j];
    }
    #pragma unroll
    for (int j = 0; j < 4; ++j) {
        int n = nb + j;
        float lm = logf(fmaxf(mask[b * NN + n], 1e-6f));
        float sc = fminf(fmaxf(s[j] * (1.0f / 16.0f) + lm, -50.0f), 50.0f) * 2.0f;
        scores[b * NN + n] = fminf(fmaxf(sc, -50.0f), 50.0f);
    }
}

// ---------------------------------------------------------------------------
// step: softmax + centroid + entropy + mask update + map output
// ---------------------------------------------------------------------------
__global__ __launch_bounds__(256) void k_softmax(const float* __restrict__ scores,
                                                 float* __restrict__ attn,
                                                 float* __restrict__ mask,
                                                 float* __restrict__ ent,
                                                 float* __restrict__ dout, int t) {
    int b = blockIdx.x, tid = threadIdx.x;
    float l[16];
    float mx = -3.4e38f;
    #pragma unroll
    for (int i = 0; i < 16; ++i) {
        l[i] = scores[b * NN + tid + i * 256];
        mx = fmaxf(mx, l[i]);
    }
    mx = bmax256(mx);
    float se = 0.0f;
    #pragma unroll
    for (int i = 0; i < 16; ++i) { l[i] = expf(l[i] - mx); se += l[i]; }
    se = bsum256(se);
    float inv = 1.0f / se;
    float rc = 0.0f, cc = 0.0f, en = 0.0f;
    #pragma unroll
    for (int i = 0; i < 16; ++i) {
        int   n = tid + i * 256;
        float a = fmaxf(l[i] * inv, 1e-8f);
        attn[b * NN + n] = a;
        dout[OUT_MAPS + (size_t)(b * KK + t) * NN + n] = a;
        rc += a * (float)(n >> 6);
        cc += a * (float)(n & 63);
        float ac = fmaxf(a, 1e-6f);
        en -= ac * logf(ac);
        float mv = mask[b * NN + n] * (1.0f - 0.9f * a);
        mask[b * NN + n] = fmaxf(mv, 1e-6f);
    }
    rc = bsum256(rc);
    cc = bsum256(cc);
    en = bsum256(en);
    if (tid == 0) {
        dout[(b * KK + t) * 2 + 0] = rc;
        dout[(b * KK + t) * 2 + 1] = cc;
        ent[b] = en / logf(4096.0f + 1e-6f);
    }
}

// ---------------------------------------------------------------------------
// step: attended[b,d] = sum_n attn[b,n] * vp[b,d,n]
// ---------------------------------------------------------------------------
__global__ __launch_bounds__(256) void k_attended(const _Float16* __restrict__ vp,
                                                  const float* __restrict__ attn,
                                                  float* __restrict__ att) {
    int b  = blockIdx.x >> 3;
    int dc = (blockIdx.x & 7) * 32;
    __shared__ float aw[NN];                         // 16 KB
    for (int i = threadIdx.x; i < NN; i += 256) aw[i] = attn[b * NN + i];
    __syncthreads();
    int nb = threadIdx.x * 16;
    for (int dd = 0; dd < 32; ++dd) {
        int d = dc + dd;
        const _Float16* vr = vp + (size_t)(b * DD + d) * NN + nb;
        v8h x0 = *(const v8h*)vr;
        v8h x1 = *(const v8h*)(vr + 8);
        float p = 0.0f;
        #pragma unroll
        for (int j = 0; j < 8; ++j) {
            p += aw[nb + j]     * (float)x0[j];
            p += aw[nb + 8 + j] * (float)x1[j];
        }
        p = bsum256(p);
        if (threadIdx.x == 0) att[b * DD + d] = p;
    }
}

// ---------------------------------------------------------------------------
// step: GRU cell + stop head
// ---------------------------------------------------------------------------
__global__ __launch_bounds__(256) void k_update(const float* __restrict__ att,
                                                const float* __restrict__ ent,
                                                float* __restrict__ state,
                                                const float* __restrict__ wih,
                                                const float* __restrict__ bih,
                                                const float* __restrict__ whh,
                                                const float* __restrict__ bhh,
                                                const float* __restrict__ sw1,
                                                const float* __restrict__ sb1,
                                                const float* __restrict__ sw2,
                                                const float* __restrict__ sb2,
                                                float* __restrict__ dout, int t) {
    int b = blockIdx.x, tid = threadIdx.x;
    __shared__ float xa[DD], hs[DD], s1[128];
    xa[tid] = att[b * DD + tid];
    hs[tid] = state[b * DD + tid];
    __syncthreads();

    float gir = bih[tid], giz = bih[DD + tid], gin = bih[2 * DD + tid];
    float ghr = bhh[tid], ghz = bhh[DD + tid], ghn = bhh[2 * DD + tid];
    for (int d = 0; d < DD; ++d) {
        float x = xa[d], h = hs[d];
        gir += x * wih[(size_t)tid * DD + d];
        giz += x * wih[(size_t)(DD + tid) * DD + d];
        gin += x * wih[(size_t)(2 * DD + tid) * DD + d];
        ghr += h * whh[(size_t)tid * DD + d];
        ghz += h * whh[(size_t)(DD + tid) * DD + d];
        ghn += h * whh[(size_t)(2 * DD + tid) * DD + d];
    }
    float rr = 1.0f / (1.0f + expf(-(gir + ghr)));
    float zz = 1.0f / (1.0f + expf(-(giz + ghz)));
    float nn = tanhf(gin + rr * ghn);
    state[b * DD + tid] = (1.0f - zz) * nn + zz * hs[tid];

    float e = ent[b];
    if (tid < 128) {
        float a = sb1[tid];
        for (int d = 0; d < DD; ++d) a += xa[d] * sw1[tid * 257 + d];
        a += e * sw1[tid * 257 + 256];
        s1[tid] = gelu_exact(a);
    }
    __syncthreads();
    if (tid == 0) {
        float raw = sb2[0];
        for (int j = 0; j < 128; ++j) raw += s1[j] * sw2[j];
        dout[OUT_STOP + b * KK + t] = 4.0f * tanhf(raw * 0.25f);
    }
}

// ---------------------------------------------------------------------------
// host launcher
// ---------------------------------------------------------------------------
extern "C" void kernel_launch(void* const* d_in, const int* in_sizes, int n_in,
                              void* d_out, int out_size, void* d_ws, size_t ws_size,
                              hipStream_t stream) {
    const float* features = (const float*)d_in[0];
    const float* task_ctx = (const float*)d_in[1];
    const float* ctx_w1   = (const float*)d_in[2];
    const float* ctx_b1   = (const float*)d_in[3];
    const float* ctx_w2   = (const float*)d_in[4];
    const float* ctx_b2   = (const float*)d_in[5];
    const float* q_w      = (const float*)d_in[6];
    const float* q_b      = (const float*)d_in[7];
    const float* k_w      = (const float*)d_in[8];
    const float* k_b      = (const float*)d_in[9];
    const float* v_w      = (const float*)d_in[10];
    const float* v_b      = (const float*)d_in[11];
    const float* qn_g     = (const float*)d_in[12];
    const float* qn_b     = (const float*)d_in[13];
    const float* fn_g     = (const float*)d_in[14];
    const float* fn_b     = (const float*)d_in[15];
    const float* stop_w1  = (const float*)d_in[16];
    const float* stop_b1  = (const float*)d_in[17];
    const float* stop_w2  = (const float*)d_in[18];
    const float* stop_b2  = (const float*)d_in[19];
    const float* gru_wih  = (const float*)d_in[20];
    const float* gru_bih  = (const float*)d_in[21];
    const float* gru_whh  = (const float*)d_in[22];
    const float* gru_bhh  = (const float*)d_in[23];

    char* ws = (char*)d_ws;
    float*     h1     = (float*)(ws + WS_H1);
    float*     clue   = (float*)(ws + WS_CLUE);
    _Float16*  wpack  = (_Float16*)(ws + WS_WPACK);
    float*     state  = (float*)(ws + WS_STATE);
    float*     mask   = (float*)(ws + WS_MASK);
    float*     qv     = (float*)(ws + WS_Q);
    float*     scores = (float*)(ws + WS_SCORES);
    float*     attn   = (float*)(ws + WS_ATTN);
    float*     att    = (float*)(ws + WS_ATT);
    float*     ent    = (float*)(ws + WS_ENT);
    _Float16*  kp     = (_Float16*)(ws + WS_KP);
    _Float16*  vp     = (_Float16*)(ws + WS_VP);
    float*     dout   = (float*)d_out;

    k_init<<<512, 256, 0, stream>>>(state, mask);
    k_ctx1<<<NB, 256, 0, stream>>>(task_ctx, ctx_w1, ctx_b1, h1);
    k_ctx2<<<dim3(NB, KK), 256, 0, stream>>>(h1, ctx_w2, ctx_b2, clue);
    k_pack<<<32, 256, 0, stream>>>(k_w, v_w, wpack);
    k_lnkv<<<NB * (NN / TM), 256, 0, stream>>>(features, fn_g, fn_b, wpack,
                                               k_b, v_b, kp, vp);

    for (int t = 0; t < KK; ++t) {
        k_query<<<NB, 256, 0, stream>>>(clue, state, qn_g, qn_b, q_w, q_b, qv, t);
        k_scores<<<NB * 4, 256, 0, stream>>>(kp, qv, mask, scores);
        k_softmax<<<NB, 256, 0, stream>>>(scores, attn, mask, ent, dout, t);
        k_attended<<<NB * 8, 256, 0, stream>>>(vp, attn, att);
        k_update<<<NB, 256, 0, stream>>>(att, ent, state,
                                         gru_wih, gru_bih, gru_whh, gru_bhh,
                                         stop_w1, stop_b1, stop_w2, stop_b2,
                                         dout, t);
    }
}
